// CompactnessLoss_29446295781542
// MI455X (gfx1250) — 8-bit matrix operand order repaired and checked by reading, compiled, not run on hardware
//
#include <hip/hip_runtime.h>

#define DD 64
#define HH 256
#define WW 256
#define NVOX (DD * HH * WW)
#define GTBL 4096
#define LTBL 1024
#define FOUR_PI 12.566370614359172f
#define EPSF 1e-5f

typedef __attribute__((ext_vector_type(8))) int v8i;

__device__ __forceinline__ int refl(int i, int n) {
  if (i < 0) return -i;
  if (i >= n) return 2 * n - 2 - i;
  return i;
}

__device__ int findRoot(int* P, int x) {
  int p = P[x];
  while (p != x) {
    int gp = P[p];
    if (gp != p) P[x] = gp;
    x = p; p = gp;
  }
  return x;
}

__device__ void unite(int* P, int a, int b) {
  a = findRoot(P, a);
  b = findRoot(P, b);
  while (a != b) {
    if (a > b) { int t = a; a = b; b = t; }
    int old = atomicCAS(&P[b], b, a);
    if (old == b) return;
    b = findRoot(P, old);
    a = findRoot(P, a);
  }
}

__device__ int gInsert(int* keys, int root) {
  unsigned s = (((unsigned)root * 2654435761u) >> 8) & (GTBL - 1);
  for (int p = 0; p < GTBL; ++p) {
    int prev = atomicCAS(&keys[s], -1, root);
    if (prev == -1 || prev == root) return (int)s;
    s = (s + 1) & (GTBL - 1);
  }
  return -1;
}

__device__ int lInsert(int* keys, int root) {
  unsigned s = (((unsigned)root * 2654435761u) >> 8) & (LTBL - 1);
  for (int p = 0; p < LTBL; ++p) {
    int prev = atomicCAS(&keys[s], -1, root);
    if (prev == -1 || prev == root) return (int)s;
    s = (s + 1) & (LTBL - 1);
  }
  return -1;
}

__global__ void k_init(const float* __restrict__ yp, int* __restrict__ parent,
                       int* __restrict__ gkeys, unsigned* __restrict__ garea,
                       unsigned* __restrict__ gperim) {
  int i = blockIdx.x * blockDim.x + threadIdx.x;
  if (i < NVOX) parent[i] = (yp[i] > 0.5f) ? i : -1;
  if (i < GTBL) { gkeys[i] = -1; garea[i] = 0u; gperim[i] = 0u; }
}

__global__ void k_link(int* __restrict__ parent) {
  int i = blockIdx.x * blockDim.x + threadIdx.x;
  if (i >= NVOX) return;
  if (parent[i] < 0) return;
  int x = i & (WW - 1), y = (i >> 8) & (HH - 1), z = i >> 16;
  const int off[13][3] = {
    {1,-1,-1},{1,-1,0},{1,-1,1},{1,0,-1},{1,0,0},{1,0,1},{1,1,-1},{1,1,0},{1,1,1},
    {0,1,-1},{0,1,0},{0,1,1},{0,0,1}
  };
#pragma unroll
  for (int o = 0; o < 13; ++o) {
    int zz = z + off[o][0], yy = y + off[o][1], xx = x + off[o][2];
    if ((unsigned)zz < (unsigned)DD && (unsigned)yy < (unsigned)HH &&
        (unsigned)xx < (unsigned)WW) {
      int nidx = (zz << 16) + (yy << 8) + xx;
      if (parent[nidx] >= 0) unite(parent, i, nidx);
    }
  }
}

__global__ void k_flatten(int* __restrict__ parent) {
  int i = blockIdx.x * blockDim.x + threadIdx.x;
  if (i >= NVOX) return;
  if (parent[i] >= 0) parent[i] = findRoot(parent, i);
}

__device__ __forceinline__ unsigned pack4(const unsigned char* h, int base,
                                          int k0, int kmax) {
  unsigned r = 0;
#pragma unroll
  for (int j = 0; j < 4; ++j) {
    int k = k0 + j;
    unsigned v = (k <= kmax) ? (unsigned)h[base + k] : 0u;
    r |= v << (8 * j);
  }
  return r;
}

__device__ v8i makeA(const unsigned char* h, int lane, int dz, int dy) {
  int m = lane & 15;
  int base = ((dz + 1) * 18 + (m + dy + 1)) * 18;
  v8i a = {};
  if (lane < 16) {
    a[0] = (int)pack4(h, base, 0, 17);
    a[1] = (int)pack4(h, base, 4, 17);
    a[2] = (int)pack4(h, base, 8, 17);
    a[3] = (int)pack4(h, base, 12, 17);
  } else {
    a[0] = (int)pack4(h, base, 16, 17);
  }
  return a;
}

__device__ v8i makeB(int lane, unsigned mask3) {
  int n = lane & 15;
  int kb = (lane < 16) ? 0 : 16;
  v8i b = {};
#pragma unroll
  for (int v = 0; v < 4; ++v) {
    unsigned w = 0;
#pragma unroll
    for (int j = 0; j < 4; ++j) {
      int k = kb + v * 4 + j;
      int d = k - n - 1;
      unsigned bit = (d >= -1 && d <= 1) ? ((mask3 >> (d + 1)) & 1u) : 0u;
      w |= bit << (8 * j);
    }
    b[v] = (int)w;
  }
  return b;
}

__global__ void __launch_bounds__(256) k_count(
    const float* __restrict__ yp, const int* __restrict__ parent,
    int* __restrict__ gkeys, unsigned* __restrict__ garea,
    unsigned* __restrict__ gperim) {
  __shared__ unsigned char halo[8][1024];
  __shared__ int lkeys[LTBL];
  __shared__ unsigned larea[LTBL];
  __shared__ unsigned lperim[LTBL];

  const int tid  = threadIdx.x;
  const int lane = tid & 31;
  const int wave = tid >> 5;

  for (int s = tid; s < LTBL; s += 256) { lkeys[s] = -1; larea[s] = 0u; lperim[s] = 0u; }
  __syncthreads();

  const int tileId = blockIdx.x * 8 + wave;
  const int z0  = tileId >> 8;
  const int rem = tileId & 255;
  const int y0  = (rem >> 4) << 4;
  const int x0  = (rem & 15) << 4;

  unsigned char* h = halo[wave];
  for (int t = lane; t < 972; t += 32) {
    int hz = t / 324, r2 = t - hz * 324;
    int hy = r2 / 18, hx = r2 - hy * 18;
    int gz = refl(z0 + hz - 1, DD);
    int gy = refl(y0 + hy - 1, HH);
    int gx = refl(x0 + hx - 1, WW);
    h[t] = (yp[(gz << 16) + (gy << 8) + gx] > 0.5f) ? 1 : 0;
  }
  __syncthreads();

  const v8i B_S0 = makeB(lane, 0x2u);
  const v8i B_S1 = makeB(lane, 0x5u);
  const v8i B_S2 = makeB(lane, 0x7u);

  v8i c = {};
#pragma unroll
  for (int dz = -1; dz <= 1; ++dz) {
#pragma unroll
    for (int dy = -1; dy <= 1; ++dy) {
      int nz = (dz != 0) + (dy != 0);
      v8i a = makeA(h, lane, dz, dy);
      v8i b = (nz == 2) ? B_S0 : (nz == 1) ? B_S2 : B_S1;
      c = __builtin_amdgcn_wmma_i32_16x16x64_iu8(false, a, false, b, c,
                                                 false, false);
    }
  }
  const int n = lane & 15;
  const int mb = (lane >> 4) * 8;
#pragma unroll
  for (int j = 0; j < 8; ++j) {
    int gy = y0 + mb + j, gx = x0 + n;
    int idx = (z0 << 16) + (gy << 8) + gx;
    int root = parent[idx];
    if (root >= 0) {
      int bnd = (c[j] < 18) ? 1 : 0;
      int s = lInsert(lkeys, root);
      if (s >= 0) {
        atomicAdd(&larea[s], 1u);
        if (bnd) atomicAdd(&lperim[s], 1u);
      } else {
        int gs = gInsert(gkeys, root);
        if (gs >= 0) {
          atomicAdd(&garea[gs], 1u);
          if (bnd) atomicAdd(&gperim[gs], 1u);
        }
      }
    }
  }
  __syncthreads();
  for (int s = tid; s < LTBL; s += 256) {
    int key = lkeys[s];
    if (key != -1) {
      int gs = gInsert(gkeys, key);
      if (gs >= 0) {
        if (larea[s])  atomicAdd(&garea[gs],  larea[s]);
        if (lperim[s]) atomicAdd(&gperim[gs], lperim[s]);
      }
    }
  }
}

__global__ void k_final(const int* __restrict__ gkeys,
                        const unsigned* __restrict__ garea,
                        const unsigned* __restrict__ gperim,
                        float* __restrict__ out) {
  __shared__ float red[256];
  float local = 0.f;
  for (int s = threadIdx.x; s < GTBL; s += 256) {
    if (gkeys[s] != -1) {
      float a = (float)garea[s];
      float p = (float)gperim[s];
      local += FOUR_PI * a / (p * p + EPSF);
    }
  }
  red[threadIdx.x] = local;
  __syncthreads();
  for (int off = 128; off > 0; off >>= 1) {
    if (threadIdx.x < off) red[threadIdx.x] += red[threadIdx.x + off];
    __syncthreads();
  }
  if (threadIdx.x == 0) {
    float comp = red[0] / (float)DD;
    out[0] = 1.0f / (comp + EPSF);
  }
}

extern "C" void kernel_launch(void* const* d_in, const int* in_sizes, int n_in,
                              void* d_out, int out_size, void* d_ws,
                              size_t ws_size, hipStream_t stream) {
  (void)in_sizes; (void)n_in; (void)out_size; (void)ws_size;
  const float* yp = (const float*)d_in[0];
  float* out = (float*)d_out;

  char* ws = (char*)d_ws;
  int*      parent = (int*)ws;
  int*      gkeys  = (int*)(ws + (size_t)NVOX * 4);
  unsigned* garea  = (unsigned*)(ws + (size_t)NVOX * 4 + GTBL * 4);
  unsigned* gperim = (unsigned*)(ws + (size_t)NVOX * 4 + GTBL * 8);

  const int threads = 256;
  const int blocksN = NVOX / threads;

  k_init   <<<blocksN, threads, 0, stream>>>(yp, parent, gkeys, garea, gperim);
  k_link   <<<blocksN, threads, 0, stream>>>(parent);
  k_flatten<<<blocksN, threads, 0, stream>>>(parent);
  k_count  <<<16384 / 8, threads, 0, stream>>>(yp, parent, gkeys, garea, gperim);
  k_final  <<<1, threads, 0, stream>>>(gkeys, garea, gperim, out);
}
